// MultiHeadsAttention_75960791597414
// MI455X (gfx1250) — matrix operand order repaired and checked by reading, not compiled, not run on hardware
//
#include <hip/hip_runtime.h>
#include <hip/hip_bf16.h>
#include <math.h>
#include <stdint.h>

typedef __attribute__((ext_vector_type(2)))  float  v2f;
typedef __attribute__((ext_vector_type(8)))  float  v8f;
typedef __attribute__((ext_vector_type(16))) __bf16 v16bf;
typedef int v4i_ __attribute__((vector_size(4 * sizeof(int))));

#define HIDDEN   1024
#define HEADS    16
#define HEAD_DIM 64
#define BATCH    4
#define SEQ      2048

// ---------------------------------------------------------------------------
// Async Global->LDS copy, 16 bytes per lane (no VGPR round trip, ASYNCcnt).
// ---------------------------------------------------------------------------
__device__ __forceinline__ void async_b128(const void* g, void* l) {
  __builtin_amdgcn_global_load_async_to_lds_b128(
      (__attribute__((address_space(1))) v4i_*)(uintptr_t)g,
      (__attribute__((address_space(3))) v4i_*)(unsigned)(uintptr_t)l,
      0, 0);
}
__device__ __forceinline__ void wait_async0() {
  asm volatile("s_wait_asynccnt 0x0" ::: "memory");
}

// --- 16-lane-half reductions (wave32; C-matrix rows live inside one half) ---
__device__ __forceinline__ float half_max16(float v) {
  v = fmaxf(v, __shfl_xor(v, 1, 32));
  v = fmaxf(v, __shfl_xor(v, 2, 32));
  v = fmaxf(v, __shfl_xor(v, 4, 32));
  v = fmaxf(v, __shfl_xor(v, 8, 32));
  return v;
}
__device__ __forceinline__ float half_sum16(float v) {
  v += __shfl_xor(v, 1, 32);
  v += __shfl_xor(v, 2, 32);
  v += __shfl_xor(v, 4, 32);
  v += __shfl_xor(v, 8, 32);
  return v;
}

// ============================================================================
// Phase 1: C = X*W + b with fused RoPE epilogue; async-staged, double-buffered.
//   mode 0 (Q): RoPE, f32 out [B,H,S,D]
//   mode 1 (K): RoPE, f32 out [B,H,S,D] + bf16 copy [B,H,S,D]
//   mode 2 (V): no RoPE, bf16 transposed out [B,H,D,S]
// 4 waves/block; each wave owns a 16x16 tile of a 16x64 block tile.
// fp32 WMMA 16x16x4 keeps full reference precision (q/k are outputs).
// ============================================================================
__global__ __launch_bounds__(128)
void proj_rope_kernel(const float* __restrict__ X, const float* __restrict__ W,
                      const float* __restrict__ bias,
                      float* __restrict__ out_f32, __bf16* __restrict__ out_bf,
                      int mode)
{
  __shared__ alignas(16) float lds_a[2][16 * 68];   // 16x64, row stride 272B
  __shared__ alignas(16) float lds_b[2][64 * 68];   // 64x64

  const int tid  = threadIdx.x;
  const int wave = tid >> 5;
  const int lane = tid & 31;
  const int r    = lane & 15;
  const int hl   = lane >> 4;
  const int m0   = blockIdx.x << 4;            // 16 rows of M = B*S
  const int n0   = blockIdx.y << 6;            // 64 cols of N = H*D
  const int col  = n0 + (wave << 4) + r;

  v8f acc;
  #pragma unroll
  for (int i = 0; i < 8; ++i) acc[i] = 0.0f;

  // async stage of one K-chunk (A: 16x64 f32, W: 64x64 f32) into buffer `buf`
  auto stage = [&](int buf, int k0) {
    #pragma unroll
    for (int i = tid; i < 256; i += 128) {          // A: 256 x 16B chunks
      int rr = i >> 4, cc = (i & 15) << 2;
      async_b128(X + (size_t)(m0 + rr) * HIDDEN + k0 + cc,
                 &lds_a[buf][rr * 68 + cc]);
    }
    #pragma unroll
    for (int i = tid; i < 1024; i += 128) {         // W: 1024 x 16B chunks
      int rr = i >> 4, cc = (i & 15) << 2;
      async_b128(W + (size_t)(k0 + rr) * HIDDEN + n0 + cc,
                 &lds_b[buf][rr * 68 + cc]);
    }
  };

  stage(0, 0);
  const int bcol = (wave << 4) + r;
  for (int kb = 0; kb < HIDDEN / 64; ++kb) {
    const int p = kb & 1;
    wait_async0();              // this wave's copies into buf p complete
    __syncthreads();            // all waves done (also fences prior reads of p^1)
    if (kb + 1 < HIDDEN / 64) stage(p ^ 1, (kb + 1) * 64);

    const float* la = lds_a[p];
    const float* lb = lds_b[p];
    #pragma unroll
    for (int kk = 0; kk < 64; kk += 4) {
      // A 16x4: lanes0-15 {K0,K1}, lanes16-31 {K2,K3}; row = lane&15
      v2f a, b;
      a[0] = la[r * 68 + kk + 2 * hl];
      a[1] = la[r * 68 + kk + 2 * hl + 1];
      // B 4x16 (mirrors A): VGPR0=K{0|2}, VGPR1=K{1|3}; col = lane&15
      b[0] = lb[(kk + 2 * hl) * 68 + bcol];
      b[1] = lb[(kk + 2 * hl + 1) * 68 + bcol];
      acc = __builtin_amdgcn_wmma_f32_16x16x4_f32(false, a, false, b,
                                                  (short)0, acc, false, false);
    }
    __syncthreads();            // reads of buf p done before next stage into p
  }

  const int head  = col >> 6;
  const int d     = col & 63;
  const int bidx  = m0 >> 11;                  // S = 2048 rows per batch
  const int sbase = m0 & (SEQ - 1);
  const float bval = bias[col];
  // inv_freq = 10000^(-(d/2)/32) = exp(-(d/2) * ln(10000)/32)
  const float inv = (mode != 2) ? __expf(-(float)(d >> 1) * 0.28782313662425572f) : 0.0f;

  #pragma unroll
  for (int i = 0; i < 8; ++i) {
    const int s = sbase + i + 8 * hl;          // C layout: VGPR i -> row i (+8 upper half)
    float v = acc[i] + bval;
    if (mode != 2) {
      // rotate-half: even d -> -x[d+1], odd d -> x[d-1]; partner in lane^1
      float part = __shfl_xor(v, 1, 32);
      float rot  = (d & 1) ? part : -part;
      float ang  = (float)s * inv;
      v = v * __cosf(ang) + rot * __sinf(ang);
      const size_t idx = ((size_t)(bidx * HEADS + head) * SEQ + s) * HEAD_DIM + d;
      out_f32[idx] = v;
      if (mode == 1) out_bf[idx] = (__bf16)v;
    } else {
      out_bf[((size_t)(bidx * HEADS + head) * HEAD_DIM + d) * SEQ + s] = (__bf16)v;
    }
  }
}

// ============================================================================
// Phase 2: flash attention per (b,h). Block = 4 waves x 16 query rows.
// Key blocks of 32; bf16 WMMA 16x16x32; K/V^T tiles async-copied (bf16, raw)
// straight into double-buffered LDS; prefetch overlaps WMMA+softmax.
// ============================================================================
#define KSTRIDE 72    // 144B rows: 16B-aligned async writes, conflict-free reads
#define VSTRIDE 40    // 80B rows
#define PSTRIDE 40

__global__ __launch_bounds__(128)
void flash_attn_kernel(const float* __restrict__ Q, const __bf16* __restrict__ Kbf,
                       const __bf16* __restrict__ Vtbf, const float* __restrict__ mask,
                       float* __restrict__ ctx)
{
  __shared__ alignas(16) __bf16 lds_k[2][32 * KSTRIDE];   // [32 keys][64 d]
  __shared__ alignas(16) __bf16 lds_v[2][64 * VSTRIDE];   // [64 d][32 keys]
  __shared__ alignas(16) __bf16 lds_p[4][16 * PSTRIDE];   // per-wave P tile

  const int tid  = threadIdx.x;
  const int wave = tid >> 5;
  const int lane = tid & 31;
  const int r    = lane & 15;
  const int hl   = lane >> 4;
  const int bh   = blockIdx.y;                 // b*H + h
  const int b    = bh >> 4;
  const int head = bh & 15;
  const int q0   = (blockIdx.x << 6) + (wave << 4);

  // Q A-fragments (16x32 bf16): element e -> K = e + 16*half
  v16bf aq0, aq1;
  {
    const size_t qrow = ((size_t)bh * SEQ + q0 + r) * HEAD_DIM;
    #pragma unroll
    for (int e = 0; e < 16; ++e) {
      const int kk = e + 16 * hl;
      aq0[e] = (__bf16)Q[qrow + kk];
      aq1[e] = (__bf16)Q[qrow + 32 + kk];
    }
  }

  v8f acc0, acc1, acc2, acc3;
  #pragma unroll
  for (int i = 0; i < 8; ++i) { acc0[i] = 0; acc1[i] = 0; acc2[i] = 0; acc3[i] = 0; }
  float mrow[8], lrow[8];
  #pragma unroll
  for (int i = 0; i < 8; ++i) { mrow[i] = -3.0e38f; lrow[i] = 0.0f; }

  const float scale = 0.03125f;                // 1/sqrt(H*D) = 1/sqrt(1024)
  __bf16* pw = lds_p[wave];

  auto stage = [&](int buf, int c0) {
    #pragma unroll
    for (int i = tid; i < 256; i += 128) {     // K: 32 rows x 8 chunks of 8 bf16
      int rr = i >> 3, cc = (i & 7) << 3;
      async_b128(Kbf + ((size_t)bh * SEQ + c0 + rr) * HEAD_DIM + cc,
                 &lds_k[buf][rr * KSTRIDE + cc]);
    }
    #pragma unroll
    for (int i = tid; i < 256; i += 128) {     // V^T: 64 rows x 4 chunks of 8 bf16
      int rr = i >> 2, cc = (i & 3) << 3;
      async_b128(Vtbf + ((size_t)bh * HEAD_DIM + rr) * SEQ + c0 + cc,
                 &lds_v[buf][rr * VSTRIDE + cc]);
    }
  };

  stage(0, 0);
  for (int n = 0; n < SEQ / 32; ++n) {
    const int c0 = n << 5;
    const int p  = n & 1;
    wait_async0();
    __syncthreads();                           // buf p ready; prior reads of p^1 done
    if (c0 + 32 < SEQ) stage(p ^ 1, c0 + 32);  // prefetch next block

    const __bf16* lk = lds_k[p];
    const __bf16* lv = lds_v[p];

    // scores: S[16q][32k] = Q(16x64) . K_blk(32x64)^T
    v8f s0, s1;
    #pragma unroll
    for (int i = 0; i < 8; ++i) { s0[i] = 0; s1[i] = 0; }
    {
      v16bf bk;   // B frag: element e -> K(=d) = e + 16*half, col = key = lane&15
      #pragma unroll
      for (int e = 0; e < 16; ++e) bk[e] = lk[r * KSTRIDE + e + 16 * hl];
      s0 = __builtin_amdgcn_wmma_f32_16x16x32_bf16(false, aq0, false, bk, (short)0, s0, false, false);
      #pragma unroll
      for (int e = 0; e < 16; ++e) bk[e] = lk[r * KSTRIDE + 32 + e + 16 * hl];
      s0 = __builtin_amdgcn_wmma_f32_16x16x32_bf16(false, aq1, false, bk, (short)0, s0, false, false);
      #pragma unroll
      for (int e = 0; e < 16; ++e) bk[e] = lk[(16 + r) * KSTRIDE + e + 16 * hl];
      s1 = __builtin_amdgcn_wmma_f32_16x16x32_bf16(false, aq0, false, bk, (short)0, s1, false, false);
      #pragma unroll
      for (int e = 0; e < 16; ++e) bk[e] = lk[(16 + r) * KSTRIDE + 32 + e + 16 * hl];
      s1 = __builtin_amdgcn_wmma_f32_16x16x32_bf16(false, aq1, false, bk, (short)0, s1, false, false);
    }

    const float m0v = mask[(size_t)b * SEQ + c0 + r];
    const float m1v = mask[(size_t)b * SEQ + c0 + 16 + r];

    // online softmax per row (row i+8*hl lives inside the lane's 16-lane half)
    #pragma unroll
    for (int i = 0; i < 8; ++i) {
      float x0 = s0[i] * scale + m0v;
      float x1 = s1[i] * scale + m1v;
      float t  = half_max16(fmaxf(x0, x1));
      float mn = fmaxf(mrow[i], t);
      float alpha = __expf(mrow[i] - mn);
      mrow[i] = mn;
      float p0 = __expf(x0 - mn);
      float p1 = __expf(x1 - mn);
      lrow[i] = lrow[i] * alpha + half_sum16(p0 + p1);
      acc0[i] *= alpha; acc1[i] *= alpha; acc2[i] *= alpha; acc3[i] *= alpha;
      pw[(i + 8 * hl) * PSTRIDE + r]      = (__bf16)p0;   // C-layout -> LDS
      pw[(i + 8 * hl) * PSTRIDE + 16 + r] = (__bf16)p1;
    }
    asm volatile("" ::: "memory");   // keep LDS P stores before A-frag reloads

    v16bf ap;                        // P as A-matrix fragment (16x32 bf16)
    #pragma unroll
    for (int e = 0; e < 16; ++e) {
      const int kk = e + 16 * hl;
      ap[e] = pw[r * PSTRIDE + kk];
    }
    {
      v16bf bv;                      // V B-frag: K = key = e + 16*half, col = d
      #pragma unroll
      for (int e = 0; e < 16; ++e) bv[e] = lv[(0 * 16 + r) * VSTRIDE + e + 16 * hl];
      acc0 = __builtin_amdgcn_wmma_f32_16x16x32_bf16(false, ap, false, bv, (short)0, acc0, false, false);
      #pragma unroll
      for (int e = 0; e < 16; ++e) bv[e] = lv[(1 * 16 + r) * VSTRIDE + e + 16 * hl];
      acc1 = __builtin_amdgcn_wmma_f32_16x16x32_bf16(false, ap, false, bv, (short)0, acc1, false, false);
      #pragma unroll
      for (int e = 0; e < 16; ++e) bv[e] = lv[(2 * 16 + r) * VSTRIDE + e + 16 * hl];
      acc2 = __builtin_amdgcn_wmma_f32_16x16x32_bf16(false, ap, false, bv, (short)0, acc2, false, false);
      #pragma unroll
      for (int e = 0; e < 16; ++e) bv[e] = lv[(3 * 16 + r) * VSTRIDE + e + 16 * hl];
      acc3 = __builtin_amdgcn_wmma_f32_16x16x32_bf16(false, ap, false, bv, (short)0, acc3, false, false);
    }
    // no trailing barrier: next iteration's top barrier fences reads of buf p
  }

  // normalize + store ctx [B,S,H*D]
  #pragma unroll
  for (int i = 0; i < 8; ++i) {
    const float invl = 1.0f / lrow[i];
    const size_t row = ((size_t)b * SEQ + q0 + i + 8 * hl) * (HEADS * HEAD_DIM)
                     + (size_t)head * HEAD_DIM;
    ctx[row +  0 + r] = acc0[i] * invl;
    ctx[row + 16 + r] = acc1[i] * invl;
    ctx[row + 32 + r] = acc2[i] * invl;
    ctx[row + 48 + r] = acc3[i] * invl;
  }
}

extern "C" void kernel_launch(void* const* d_in, const int* in_sizes, int n_in,
                              void* d_out, int out_size, void* d_ws, size_t ws_size,
                              hipStream_t stream) {
  const float* X    = (const float*)d_in[0];
  const float* Wq   = (const float*)d_in[1];
  const float* bq   = (const float*)d_in[2];
  const float* Wk   = (const float*)d_in[3];
  const float* bk   = (const float*)d_in[4];
  const float* Wv   = (const float*)d_in[5];
  const float* bv   = (const float*)d_in[6];
  const float* mask = (const float*)d_in[7];
  (void)in_sizes; (void)n_in; (void)out_size; (void)ws_size;

  const size_t per = (size_t)BATCH * SEQ * HEADS * HEAD_DIM;   // 8M elements
  float*  ctx  = (float*)d_out;            // [B,S,H*D]
  float*  qout = ctx + per;                // [B,H,S,D] RoPE'd (returned)
  float*  kout = qout + per;               // [B,H,S,D] RoPE'd (returned)
  __bf16* kbf  = (__bf16*)d_ws;            // [B,H,S,D] bf16 copy of K (16 MB)
  __bf16* vtbf = kbf + per;                // [B,H,D,S] bf16 V transposed (16 MB)

  dim3 blk(128);
  dim3 g1((BATCH * SEQ) / 16, (HEADS * HEAD_DIM) / 64);
  hipLaunchKernelGGL(proj_rope_kernel, g1, blk, 0, stream, X, Wq, bq, qout, (__bf16*)nullptr, 0);
  hipLaunchKernelGGL(proj_rope_kernel, g1, blk, 0, stream, X, Wk, bk, kout, kbf, 1);
  hipLaunchKernelGGL(proj_rope_kernel, g1, blk, 0, stream, X, Wv, bv, (float*)nullptr, vtbf, 2);

  dim3 g2(SEQ / 64, BATCH * HEADS);
  hipLaunchKernelGGL(flash_attn_kernel, g2, blk, 0, stream, qout, kbf, vtbf, mask, ctx);
}
